// GraphModel_19851338842193
// MI455X (gfx1250) — compile-verified, matrix operand order checked by reading, not run on hardware
//
#include <hip/hip_runtime.h>
#include <hip/hip_bf16.h>
#include <math.h>

typedef __bf16 v16bf __attribute__((ext_vector_type(16)));
typedef float  v8f   __attribute__((ext_vector_type(8)));

#define GAT_ALPHA 0.2f
#define BN_EPS    1e-5f
#define MASK_NEG  (-1e9f)

#define NN   100   // nodes
#define NPAD 112   // nodes padded to 16
#define DP   128   // feature dim padded
#define HH   4     // heads
#define OO   32    // per-head out dim
#define LDA  136   // LDS halfword stride: 272B = 17*16B (aligned, conflict-free)
#define LDG  40    // GEMM tile stride: 80B = 5*16B (aligned, conflict-free)

// ---------------------------------------------------------------------------
// Unified WMMA bf16 fragment loader (wave32, CDNA5 ISA 7.12.2).
// For A (MxK row-major):            load_frag(base, ld, m0, k0)
// For B (KxN stored transposed):    load_frag(baseT, ld, n0, k0)
// Per-lane data is two contiguous 16B chunks -> 2x ds_load_b128.
// Requires: ld % 8 == 0, k0 % 8 == 0 (16B alignment).
// ---------------------------------------------------------------------------
__device__ __forceinline__ v16bf load_frag(const __bf16* base, int ld,
                                           int row0, int k0) {
  const int lane = threadIdx.x & 31;
  const __bf16* p = base + (size_t)(row0 + (lane & 15)) * ld
                         + (k0 + ((lane >> 4) << 3));
  union { v16bf v; uint4 q[2]; } u;
  u.q[0] = *reinterpret_cast<const uint4*>(p);
  u.q[1] = *reinterpret_cast<const uint4*>(p + 16);
  return u.v;
}

// ---------------------------------------------------------------------------
// One multi-head GAT layer; one workgroup (8 wave32) per graph.
// x:(B,N,F)  W:(H,F,O)  a:(H,2O)  Adj:(B,N,N)  out:(B,N,H*O)
// ---------------------------------------------------------------------------
__global__ __launch_bounds__(256) void gat_layer(
    const float* __restrict__ x, int F,
    const float* __restrict__ W,
    const float* __restrict__ a,
    const float* __restrict__ Adj,
    float* __restrict__ out) {
  __shared__ alignas(16) __bf16 xs[NPAD][LDA];    // x staged; reused as att
  __shared__ alignas(16) __bf16 hst[DP][LDA];     // h TRANSPOSED [col][node]
  __shared__ alignas(16) __bf16 wst[HH][OO][LDA]; // W TRANSPOSED [head][o][f]
  __shared__ alignas(16) float  adjf[NN * NN];    // adjacency, async-DMA'd
  __shared__ float  av[HH][2 * OO];
  __shared__ float  es[HH][NPAD];
  __shared__ float  ed[HH][NPAD];

  const int b    = blockIdx.x;
  const int tid  = threadIdx.x;
  const int wave = tid >> 5;
  const int lane = tid & 31;

  // ---- kick off async DMA of Adj[b] into LDS (overlaps with h-GEMM) ----
  {
    const unsigned long long gbase =
        (unsigned long long)(uintptr_t)(Adj + (size_t)b * NN * NN);
    const unsigned ldsbase = (unsigned)(uintptr_t)(&adjf[0]);
    for (int c = tid; c < (NN * NN) / 4; c += 256) {   // 2500 x 16B chunks
      unsigned goff  = (unsigned)c << 4;
      unsigned laddr = ldsbase + goff;
      asm volatile("global_load_async_to_lds_b128 %0, %1, %2"
                   :
                   : "v"(laddr), "v"(goff), "s"(gbase)
                   : "memory");
    }
  }

  // ---- stage x (zero-padded to NPADxDP) ----
  for (int idx = tid; idx < NPAD * DP; idx += 256) {
    int n = idx / DP, f = idx % DP;
    float v = (n < NN && f < F) ? x[(size_t)b * NN * F + (size_t)n * F + f] : 0.f;
    xs[n][f] = (__bf16)v;
  }
  // ---- stage W transposed, zero-padded along f ----
  for (int idx = tid; idx < HH * OO * DP; idx += 256) {
    int h = idx / (OO * DP);
    int r = idx % (OO * DP);
    int o = r / DP, f = r % DP;
    float v = (f < F) ? W[((size_t)h * F + f) * OO + o] : 0.f;
    wst[h][o][f] = (__bf16)v;
  }
  for (int idx = tid; idx < HH * 2 * OO; idx += 256)
    av[idx / (2 * OO)][idx % (2 * OO)] = a[idx];
  // zero node-pad region of hst (nodes 112..127) so att K-pad is 0*0
  for (int idx = tid; idx < DP * 16; idx += 256)
    hst[idx / 16][NPAD + (idx & 15)] = (__bf16)0.f;
  __syncthreads();

  // ---- h = x @ W  per head:  7 m-tiles x 4 heads x 2 n-tiles = 56 jobs ----
  const int ksteps = (F + 31) >> 5;
  for (int job = wave; job < 56; job += 8) {
    int mt = job >> 3, rem = job & 7;
    int h = rem >> 1, nt = rem & 1;
    v8f acc = {0.f, 0.f, 0.f, 0.f, 0.f, 0.f, 0.f, 0.f};
    for (int kk = 0; kk < ksteps; ++kk) {
      v16bf fa = load_frag(&xs[0][0], LDA, mt * 16, kk * 32);
      v16bf fb = load_frag(&wst[h][0][0], LDA, nt * 16, kk * 32);
      acc = __builtin_amdgcn_wmma_f32_16x16x32_bf16(false, fa, false, fb,
                                                    (short)0, acc, false, false);
    }
    // store transposed: hst[col][row0..row0+7] is contiguous -> 1x b128 store
    int col  = h * OO + nt * 16 + (lane & 15);
    int row0 = mt * 16 + ((lane >> 4) << 3);
    union { __bf16 e[8]; uint4 q; } s;
#pragma unroll
    for (int r = 0; r < 8; ++r) s.e[r] = (__bf16)acc[r];
    *reinterpret_cast<uint4*>(&hst[col][row0]) = s.q;
  }
  __syncthreads();

  // ---- e_src / e_dst ----
  for (int idx = tid; idx < HH * NPAD; idx += 256) {
    int h = idx / NPAD, n = idx % NPAD;
    float s = 0.f, d = 0.f;
#pragma unroll
    for (int o = 0; o < OO; ++o) {
      float hv = (float)hst[h * OO + o][n];
      s += hv * av[h][o];
      d += hv * av[h][OO + o];
    }
    es[h][n] = s;
    ed[h][n] = d;
  }
  // adjacency DMA must be complete before the softmax phase reads it
  asm volatile("s_wait_asynccnt 0x0" ::: "memory");
  __syncthreads();   // xs dead from here -> reuse as att (A operand, row-major)

  // ---- per head: masked softmax rows, then out = elu(att @ h_head) ----
  for (int h = 0; h < HH; ++h) {
    for (int n = tid; n < NPAD; n += 256) {
      if (n < NN) {
        const float* Arow = &adjf[n * NN];
        float esn = es[h][n];
        float mx = -INFINITY;
        for (int m = 0; m < NN; ++m) {
          float e = esn + ed[h][m];
          e = (e > 0.f) ? e : GAT_ALPHA * e;
          e = (Arow[m] > 0.f) ? e : MASK_NEG;
          mx = fmaxf(mx, e);
        }
        float sum = 0.f;
        for (int m = 0; m < NN; ++m) {
          float e = esn + ed[h][m];
          e = (e > 0.f) ? e : GAT_ALPHA * e;
          e = (Arow[m] > 0.f) ? e : MASK_NEG;
          sum += __expf(e - mx);
        }
        float inv = 1.f / sum;
        for (int m = 0; m < NN; ++m) {
          float e = esn + ed[h][m];
          e = (e > 0.f) ? e : GAT_ALPHA * e;
          e = (Arow[m] > 0.f) ? e : MASK_NEG;
          xs[n][m] = (__bf16)(__expf(e - mx) * inv);
        }
        for (int m = NN; m < DP; ++m) xs[n][m] = (__bf16)0.f;
      } else {
        for (int m = 0; m < DP; ++m) xs[n][m] = (__bf16)0.f;
      }
    }
    __syncthreads();

    // att(112x128) @ h_head(128x32): 7 m-tiles x 2 n-tiles = 14 jobs
    for (int job = wave; job < 14; job += 8) {
      int mt = job >> 1, nt = job & 1;
      v8f acc = {0.f, 0.f, 0.f, 0.f, 0.f, 0.f, 0.f, 0.f};
#pragma unroll
      for (int kk = 0; kk < 4; ++kk) {
        v16bf fa = load_frag(&xs[0][0], LDA, mt * 16, kk * 32);
        v16bf fb = load_frag(&hst[0][0], LDA, h * OO + nt * 16, kk * 32);
        acc = __builtin_amdgcn_wmma_f32_16x16x32_bf16(false, fa, false, fb,
                                                      (short)0, acc, false, false);
      }
      int col  = h * OO + nt * 16 + (lane & 15);
      int row0 = mt * 16 + ((lane >> 4) << 3);
#pragma unroll
      for (int r = 0; r < 8; ++r) {
        int row = row0 + r;
        if (row < NN) {
          float v = acc[r];
          v = (v > 0.f) ? v : (__expf(v) - 1.f);   // ELU
          out[(size_t)b * NN * (HH * OO) + (size_t)row * (HH * OO) + col] = v;
        }
      }
    }
    __syncthreads();
  }
}

// ---------------------------------------------------------------------------
// Tiled GEMM (M,K)x(K,N) + bias + eval-BN + ReLU.  128x128 block tile,
// 8 wave32; each wave owns a 32x64 sub-tile (2x4 WMMA accumulators).
// Assumes M%128==0, N%128==0, K%32==0 (true for both projection layers).
// ---------------------------------------------------------------------------
__global__ __launch_bounds__(256) void gemm_bn_relu(
    const float* __restrict__ Am,   // M x K
    const float* __restrict__ Bm,   // K x N
    const float* __restrict__ bias,
    const float* __restrict__ gam,
    const float* __restrict__ bet,
    float* __restrict__ out, int M, int Nn, int K) {
  __shared__ alignas(16) __bf16 As[128][LDG];   // A tile 128x32, row-major
  __shared__ alignas(16) __bf16 Bt[128][LDG];   // B tile 32x128, TRANSPOSED [n][k]

  const int tid  = threadIdx.x;
  const int wave = tid >> 5;
  const int lane = tid & 31;
  const int wm = wave >> 1;        // 0..3 -> 32-row strip
  const int wn = wave & 1;         // 0..1 -> 64-col strip
  const int rowBlk = blockIdx.y * 128;
  const int colBlk = blockIdx.x * 128;

  v8f acc[2][4];
#pragma unroll
  for (int i = 0; i < 2; ++i)
#pragma unroll
    for (int j = 0; j < 4; ++j)
      acc[i][j] = (v8f){0.f, 0.f, 0.f, 0.f, 0.f, 0.f, 0.f, 0.f};

  for (int k0 = 0; k0 < K; k0 += 32) {
    // prefetch next K tile into cache (global_prefetch_b8)
    if (k0 + 32 < K) {
      __builtin_prefetch(Am + (size_t)(rowBlk + (tid & 127)) * K + k0 + 32, 0, 1);
      __builtin_prefetch(Bm + (size_t)(k0 + 32 + (tid >> 3)) * Nn + colBlk
                            + ((tid & 7) << 4), 0, 1);
    }
    // stage A: float4 loads, packed 4xbf16 stores
    for (int idx = tid; idx < 128 * 8; idx += 256) {
      int r = idx >> 3, cq = (idx & 7) << 2;
      float4 v = *reinterpret_cast<const float4*>(
          Am + (size_t)(rowBlk + r) * K + k0 + cq);
      union { __bf16 e[4]; uint2 q; } s;
      s.e[0] = (__bf16)v.x; s.e[1] = (__bf16)v.y;
      s.e[2] = (__bf16)v.z; s.e[3] = (__bf16)v.w;
      *reinterpret_cast<uint2*>(&As[r][cq]) = s.q;
    }
    // stage B transposed: coalesced float4 global reads, scattered LDS stores
    for (int idx = tid; idx < 32 * 32; idx += 256) {
      int r = idx >> 5, cq = (idx & 31) << 2;
      float4 v = *reinterpret_cast<const float4*>(
          Bm + (size_t)(k0 + r) * Nn + colBlk + cq);
      Bt[cq + 0][r] = (__bf16)v.x;
      Bt[cq + 1][r] = (__bf16)v.y;
      Bt[cq + 2][r] = (__bf16)v.z;
      Bt[cq + 3][r] = (__bf16)v.w;
    }
    __syncthreads();

    v16bf fb[4];
#pragma unroll
    for (int ni = 0; ni < 4; ++ni)
      fb[ni] = load_frag(&Bt[0][0], LDG, wn * 64 + ni * 16, 0);
#pragma unroll
    for (int mi = 0; mi < 2; ++mi) {
      v16bf fa = load_frag(&As[0][0], LDG, wm * 32 + mi * 16, 0);
#pragma unroll
      for (int ni = 0; ni < 4; ++ni)
        acc[mi][ni] = __builtin_amdgcn_wmma_f32_16x16x32_bf16(
            false, fa, false, fb[ni], (short)0, acc[mi][ni], false, false);
    }
    __syncthreads();
  }

  const float scale = rsqrtf(1.0f + BN_EPS);
#pragma unroll
  for (int mi = 0; mi < 2; ++mi) {
#pragma unroll
    for (int ni = 0; ni < 4; ++ni) {
      int col  = colBlk + wn * 64 + ni * 16 + (lane & 15);
      int row0 = rowBlk + wm * 32 + mi * 16 + ((lane >> 4) << 3);
      float bb = bias[col], gg = gam[col], ee = bet[col];
#pragma unroll
      for (int r = 0; r < 8; ++r) {
        float y = acc[mi][ni][r] + bb;
        y = gg * y * scale + ee;
        out[(size_t)(row0 + r) * Nn + col] = fmaxf(y, 0.f);
      }
    }
  }
}

// ---------------------------------------------------------------------------
extern "C" void kernel_launch(void* const* d_in, const int* in_sizes, int n_in,
                              void* d_out, int out_size, void* d_ws, size_t ws_size,
                              hipStream_t stream) {
  (void)in_sizes; (void)n_in; (void)out_size; (void)ws_size;
  const float* X   = (const float*)d_in[0];
  const float* Adj = (const float*)d_in[1];
  const float* W0  = (const float*)d_in[2];
  const float* a0  = (const float*)d_in[3];
  const float* Wl  = (const float*)d_in[4];
  const float* al  = (const float*)d_in[5];
  const float* pW1 = (const float*)d_in[6];
  const float* pb1 = (const float*)d_in[7];
  const float* g1  = (const float*)d_in[8];
  const float* be1 = (const float*)d_in[9];
  const float* pW2 = (const float*)d_in[10];
  const float* pb2 = (const float*)d_in[11];
  const float* g2  = (const float*)d_in[12];
  const float* be2 = (const float*)d_in[13];
  float* outp = (float*)d_out;

  const int B = 1024, F0 = 65;
  float* buf0 = (float*)d_ws;                       // B*N*128 floats
  float* buf1 = buf0 + (size_t)B * NN * DP;         // B*N*128 floats
  float* mid  = buf1 + (size_t)B * NN * DP;         // 1024*1024 floats

  // layer 0 (F=65)
  gat_layer<<<B, 256, 0, stream>>>(X, F0, W0, a0, Adj, buf0);
  // layers 1..4 (F=128), ping-pong
  const float* cur = buf0;
  float* nxt = buf1;
  for (int i = 0; i < 4; ++i) {
    gat_layer<<<B, 256, 0, stream>>>(cur, DP,
                                     Wl + (size_t)i * HH * DP * OO,
                                     al + (size_t)i * HH * 2 * OO,
                                     Adj, nxt);
    float* t = (float*)cur; cur = nxt; nxt = t;
  }
  // cur == buf0 holds final GAT activations, laid out as (B, N*128) flat.

  dim3 grid1(1024 / 128, 1024 / 128);
  gemm_bn_relu<<<grid1, 256, 0, stream>>>(cur, pW1, pb1, g1, be1, mid,
                                          1024, 1024, 12800);
  dim3 grid2(128 / 128, 1024 / 128);
  gemm_bn_relu<<<grid2, 256, 0, stream>>>(mid, pW2, pb2, g2, be2, outp,
                                          1024, 128, 1024);
}
